// KinematicConditioner_74131135529391
// MI455X (gfx1250) — matrix operand order repaired and checked by reading, not compiled, not run on hardware
//
#include <hip/hip_runtime.h>
#include <hip/hip_bf16.h>

typedef __attribute__((ext_vector_type(16))) __bf16 v16bf;
typedef __attribute__((ext_vector_type(8)))  float  v8f;
typedef __attribute__((ext_vector_type(4)))  int    v4i;

#define C_DIM 2048
#define T_DIM 29
#define N_DIM 32
#define D_DIM 18
#define TW 4
#define TL 8
#define H_DIM 44
#define W_DIM 80
#define HW (H_DIM*W_DIM)
#define M1 (T_DIM*N_DIM)      // 928
#define MA ((TL-1)*N_DIM)     // 224
#define MG (TL*HW)            // 28160

#if __has_builtin(__builtin_amdgcn_global_load_async_to_lds_b128) && \
    __has_builtin(__builtin_amdgcn_s_wait_asynccnt)
#define HAS_ASYNC 1
#else
#define HAS_ASYNC 0
#endif

__device__ __forceinline__ float gelu_exact(float x) {
    return 0.5f * x * (1.0f + erff(x * 0.70710678118654752f));
}

#if HAS_ASYNC
typedef __attribute__((address_space(1))) v4i gv4i;   // global int4
typedef __attribute__((address_space(3))) v4i lv4i;   // LDS int4
// 16B-per-lane async copy global -> LDS (ASYNCcnt tracked, no VGPR staging)
__device__ __forceinline__ void async_ld16(const void* g, void* l) {
    __builtin_amdgcn_global_load_async_to_lds_b128((gv4i*)g, (lv4i*)l, 0, 0);
}
#endif

// ---------------------------------------------------------------------------
// Tiled bf16 WMMA GEMM:  Out(MxN) = A(MxK) @ Bt(NxK)^T  [+bias] [gelu]
// Block tile 64(M) x 256(N) x 32(K), 256 threads = 8 waves in 2x4 grid,
// each wave computes 32x64 via 2x4 v_wmma_f32_16x16x32_bf16 accumulators.
// Double-buffered LDS with GLOBAL_LOAD_ASYNC_TO_LDS_B128 staging.
// ---------------------------------------------------------------------------
template<bool GELU, bool OUT_BF16>
__global__ __launch_bounds__(256) void gemm_wmma_bf16(
    const __bf16* __restrict__ A,   // M x K row-major (lda)
    const __bf16* __restrict__ Bt,  // N x K row-major (B transposed, ldb)
    const float*  __restrict__ bias,// length N or nullptr
    void* __restrict__ Out,         // M x N (ldc), f32 or bf16
    int M, int K, int lda, int ldb, int ldc,
    long strideA, long strideB, long strideC)
{
    __shared__ __align__(32) __bf16 As[2][64 * 32];
    __shared__ __align__(32) __bf16 Bs[2][256 * 32];

    const int tid  = threadIdx.x;
    const int lane = tid & 31;
    const int wid  = tid >> 5;
    const int wm   = wid >> 2;   // 0..1  (M dir)
    const int wn   = wid & 3;    // 0..3  (N dir)

    const int bm = blockIdx.x * 64;
    const int bn = blockIdx.y * 256;
    const int bz = blockIdx.z;

    A  += (long)bz * strideA;
    Bt += (long)bz * strideB;

    v8f acc[2][4];
#pragma unroll
    for (int i = 0; i < 2; ++i)
#pragma unroll
        for (int j = 0; j < 4; ++j) {
            v8f z = {0.f,0.f,0.f,0.f,0.f,0.f,0.f,0.f};
            acc[i][j] = z;
        }

    const int arow = tid >> 2;        // 0..63
    const int acol = (tid & 3) * 8;   // 0,8,16,24
    const int half = lane >> 4;       // 0/1
    const int l16  = lane & 15;

    // Clamp instead of predicating the A-tile row: out-of-range rows load a
    // valid duplicate row (results discarded at store) -> branch-free k-loop.
    int agr = bm + arow; if (agr >= M) agr = M - 1;
    const __bf16* aSrc = A + (long)agr * lda + acol;
    const __bf16* bSrc = Bt + (long)(bn + tid) * ldb;

    const int ktiles = K >> 5;

#if HAS_ASYNC
    // prologue: async-stage tile 0 into buffer 0
    async_ld16(aSrc,      &As[0][arow * 32 + acol]);
    async_ld16(bSrc +  0, &Bs[0][tid * 32 +  0]);
    async_ld16(bSrc +  8, &Bs[0][tid * 32 +  8]);
    async_ld16(bSrc + 16, &Bs[0][tid * 32 + 16]);
    async_ld16(bSrc + 24, &Bs[0][tid * 32 + 24]);
    __builtin_amdgcn_s_wait_asynccnt(0);
    __syncthreads();

    for (int kt = 0; kt < ktiles; ++kt) {
        const int cur = kt & 1;
        const int nxt = cur ^ 1;
        if (kt + 1 < ktiles) {      // overlap next tile's HBM fetch with WMMA
            const int k1 = (kt + 1) << 5;
            async_ld16(aSrc + k1,      &As[nxt][arow * 32 + acol]);
            async_ld16(bSrc + k1 +  0, &Bs[nxt][tid * 32 +  0]);
            async_ld16(bSrc + k1 +  8, &Bs[nxt][tid * 32 +  8]);
            async_ld16(bSrc + k1 + 16, &Bs[nxt][tid * 32 + 16]);
            async_ld16(bSrc + k1 + 24, &Bs[nxt][tid * 32 + 24]);
        }
        // A fragment: contiguous K order (matches B) — lane<16 holds K {0..15},
        // lane>=16 holds K {16..31} for row M = lane%16.
        v16bf a[2], b[4];
#pragma unroll
        for (int i = 0; i < 2; ++i) {
            union { uint4 q[2]; v16bf v; } f;
            const __bf16* r = &As[cur][(wm * 32 + i * 16 + l16) * 32];
            f.q[0] = *(const uint4*)(r + half * 16);
            f.q[1] = *(const uint4*)(r + half * 16 + 8);
            a[i] = f.v;
        }
#pragma unroll
        for (int j = 0; j < 4; ++j)
            b[j] = *(const v16bf*)(&Bs[cur][(wn * 64 + j * 16 + l16) * 32 + half * 16]);
#pragma unroll
        for (int i = 0; i < 2; ++i)
#pragma unroll
            for (int j = 0; j < 4; ++j)
                acc[i][j] = __builtin_amdgcn_wmma_f32_16x16x32_bf16(
                    false, a[i], false, b[j], (short)0, acc[i][j], false, false);

        __builtin_amdgcn_s_wait_asynccnt(0);
        __syncthreads();
    }
#else
    // fallback: synchronous staging through VGPRs, single buffer
    for (int kt = 0; kt < ktiles; ++kt) {
        const int k0 = kt << 5;
        __syncthreads();
        *(uint4*)(&As[0][arow * 32 + acol]) = *(const uint4*)(aSrc + k0);
        {
            const __bf16* s = bSrc + k0;
            uint4* d = (uint4*)(&Bs[0][tid * 32]);
            d[0] = ((const uint4*)s)[0];
            d[1] = ((const uint4*)s)[1];
            d[2] = ((const uint4*)s)[2];
            d[3] = ((const uint4*)s)[3];
        }
        __syncthreads();
        v16bf a[2], b[4];
#pragma unroll
        for (int i = 0; i < 2; ++i) {
            union { uint4 q[2]; v16bf v; } f;
            const __bf16* r = &As[0][(wm * 32 + i * 16 + l16) * 32];
            f.q[0] = *(const uint4*)(r + half * 16);
            f.q[1] = *(const uint4*)(r + half * 16 + 8);
            a[i] = f.v;
        }
#pragma unroll
        for (int j = 0; j < 4; ++j)
            b[j] = *(const v16bf*)(&Bs[0][(wn * 64 + j * 16 + l16) * 32 + half * 16]);
#pragma unroll
        for (int i = 0; i < 2; ++i)
#pragma unroll
            for (int j = 0; j < 4; ++j)
                acc[i][j] = __builtin_amdgcn_wmma_f32_16x16x32_bf16(
                    false, a[i], false, b[j], (short)0, acc[i][j], false, false);
    }
#endif

    // epilogue: C/D layout — lane<16: VGPR r -> M=r, N=lane; lane>=16: M=8+r.
#pragma unroll
    for (int i = 0; i < 2; ++i) {
#pragma unroll
        for (int j = 0; j < 4; ++j) {
            int col = bn + wn * 64 + j * 16 + l16;
            float bv = bias ? bias[col] : 0.0f;
#pragma unroll
            for (int r = 0; r < 8; ++r) {
                int rt  = (lane < 16) ? r : (8 + r);
                int row = bm + wm * 32 + i * 16 + rt;
                if (row < M) {
                    float x = acc[i][j][r] + bv;
                    if (GELU) x = gelu_exact(x);
                    if (OUT_BF16) {
                        __bf16* o = (__bf16*)Out + (long)bz * strideC;
                        o[(long)row * ldc + col] = (__bf16)x;
                    } else {
                        float* o = (float*)Out + (long)bz * strideC;
                        o[(long)row * ldc + col] = x;
                    }
                }
            }
        }
    }
}

// ---------------------------------------------------------------------------
// One-time weight convert: f32 (K x N) -> bf16 transposed (N x K)
// ---------------------------------------------------------------------------
__global__ void wt_convert_kernel(const float* __restrict__ src,
                                  __bf16* __restrict__ dst, int K, int N)
{
    long idx = (long)blockIdx.x * 256 + threadIdx.x;
    if (idx >= (long)K * N) return;
    int n = (int)(idx % N);
    int k = (int)(idx / N);
    dst[(long)n * K + k] = (__bf16)src[idx];
}

// enc1: h1 = gelu(kin @ w_enc1 + b_enc1) -> bf16 (928 x 2048)
__global__ void enc1_kernel(const float* __restrict__ kin,
                            const float* __restrict__ w1,
                            const float* __restrict__ b1,
                            __bf16* __restrict__ h1)
{
    int row = blockIdx.x;
    int c   = blockIdx.y * 256 + threadIdx.x;
    __shared__ float kr[D_DIM];
    if (threadIdx.x < D_DIM) kr[threadIdx.x] = kin[row * D_DIM + threadIdx.x];
    __syncthreads();
    float s = b1[c];
#pragma unroll
    for (int d = 0; d < D_DIM; ++d) s += kr[d] * w1[d * C_DIM + c];
    h1[(long)row * C_DIM + c] = (__bf16)gelu_exact(s);
}

// pos MLP (3->128 gelu ->2 sigmoid) + validity flag, one 128-thread block/row
__global__ __launch_bounds__(128) void pos_kernel(
    const float* __restrict__ kin,
    const float* __restrict__ w1, const float* __restrict__ b1,
    const float* __restrict__ w2, const float* __restrict__ b2,
    float* __restrict__ pos, float* __restrict__ validf)
{
    int row = blockIdx.x;
    int j   = threadIdx.x;
    __shared__ float kr[D_DIM];
    __shared__ float r0[128], r1[128];
    if (j < D_DIM) kr[j] = kin[row * D_DIM + j];
    __syncthreads();
    float h = b1[j] + kr[0]*w1[j] + kr[1]*w1[128 + j] + kr[2]*w1[256 + j];
    h = gelu_exact(h);
    r0[j] = h * w2[j * 2 + 0];
    r1[j] = h * w2[j * 2 + 1];
    __syncthreads();
    for (int s = 64; s > 0; s >>= 1) {
        if (j < s) { r0[j] += r0[j + s]; r1[j] += r1[j + s]; }
        __syncthreads();
    }
    if (j == 0) {
        pos[row * 2 + 0] = 1.f / (1.f + __expf(-(r0[0] + b2[0])));
        pos[row * 2 + 1] = 1.f / (1.f + __expf(-(r1[0] + b2[1])));
        float v = kr[14] + kr[15] + kr[16] + kr[17];
        validf[row] = (v > 0.f) ? 1.f : 0.f;
    }
}

// temporal mean of emb[1:] over windows of 4 -> bf16 (224 x 2048)
__global__ void avg_kernel(const float* __restrict__ emb, __bf16* __restrict__ erin)
{
    long idx = (long)blockIdx.x * 256 + threadIdx.x;   // < 224*2048
    int c  = (int)(idx % C_DIM);
    int r  = (int)(idx / C_DIM);      // ti*32 + n
    int ti = r >> 5, n = r & 31;
    float s = 0.f;
#pragma unroll
    for (int jj = 0; jj < TW; ++jj) {
        int t = 1 + ti * TW + jj;
        s += emb[((long)(t * N_DIM + n)) * C_DIM + c];
    }
    erin[idx] = (__bf16)(s * 0.25f);
}

// e0 copy: emb rows of t=0 -> bf16 rows 0..31 of emb_w
__global__ void e0_kernel(const float* __restrict__ emb, __bf16* __restrict__ embw)
{
    long idx = (long)blockIdx.x * 256 + threadIdx.x;   // < 32*2048
    embw[idx] = (__bf16)emb[idx];
}

// window-average positions / validity -> pos_w (256x2), mask_w (256)
__global__ void posw_kernel(const float* __restrict__ pos,
                            const float* __restrict__ validf,
                            float* __restrict__ posw, float* __restrict__ maskw)
{
    int idx = threadIdx.x;           // 0..255 = tw*32 + n
    int tw = idx >> 5, n = idx & 31;
    if (tw == 0) {
        posw[idx * 2 + 0] = pos[n * 2 + 0];
        posw[idx * 2 + 1] = pos[n * 2 + 1];
        maskw[idx] = validf[n];
    } else {
        float px = 0.f, py = 0.f, m = 0.f;
#pragma unroll
        for (int jj = 0; jj < TW; ++jj) {
            int t = 1 + (tw - 1) * TW + jj;
            px += pos[(t * N_DIM + n) * 2 + 0];
            py += pos[(t * N_DIM + n) * 2 + 1];
            m = fmaxf(m, validf[t * N_DIM + n]);
        }
        posw[idx * 2 + 0] = px * 0.25f;
        posw[idx * 2 + 1] = py * 0.25f;
        maskw[idx] = m;
    }
}

// transpose emb_w per-t for use as GEMM B: embT[t][c][n] = emb_w[t*32+n][c]
__global__ void embT_kernel(const __bf16* __restrict__ embw, __bf16* __restrict__ embT)
{
    long idx = (long)blockIdx.x * 256 + threadIdx.x;   // < 256*2048
    int c = (int)(idx % C_DIM);
    int r = (int)(idx / C_DIM);       // t*32 + n
    int t = r >> 5, n = r & 31;
    embT[(long)t * C_DIM * N_DIM + (long)c * N_DIM + n] = embw[idx];
}

// normalized Gaussian splat weights -> bf16 A-matrix wnA[t][hw][n]
__global__ void wn_kernel(const float* __restrict__ posw,
                          const float* __restrict__ maskw,
                          __bf16* __restrict__ wnA)
{
    int idx = blockIdx.x * 256 + threadIdx.x;   // < 28160
    int t   = idx / HW;
    int rem = idx % HW;
    float v = (float)(rem / W_DIM) * (1.0f / (H_DIM - 1));
    float u = (float)(rem % W_DIM) * (1.0f / (W_DIM - 1));
    float wv[N_DIM];
    float s = 0.f;
#pragma unroll 8
    for (int n = 0; n < N_DIM; ++n) {
        int r = t * N_DIM + n;
        float dx = posw[r * 2 + 0] - u;
        float dy = posw[r * 2 + 1] - v;
        float ww = __expf(-(dx * dx + dy * dy) * 50.0f) * maskw[r]; // 1/(2*0.1^2)
        wv[n] = ww;
        s += ww;
    }
    float inv = 1.f / (s + 1e-8f);
    long base = (long)idx * N_DIM;
#pragma unroll 8
    for (int n = 0; n < N_DIM; ++n)
        wnA[base + n] = (__bf16)(wv[n] * inv);
}

// ---------------------------------------------------------------------------
extern "C" void kernel_launch(void* const* d_in, const int* in_sizes, int n_in,
                              void* d_out, int out_size, void* d_ws, size_t ws_size,
                              hipStream_t stream)
{
    const float* kin    = (const float*)d_in[0];
    const float* w_enc1 = (const float*)d_in[1];
    const float* b_enc1 = (const float*)d_in[2];
    const float* w_enc2 = (const float*)d_in[3];
    const float* b_enc2 = (const float*)d_in[4];
    const float* w_pos1 = (const float*)d_in[5];
    const float* b_pos1 = (const float*)d_in[6];
    const float* w_pos2 = (const float*)d_in[7];
    const float* b_pos2 = (const float*)d_in[8];
    const float* w_agg1 = (const float*)d_in[9];
    const float* b_agg1 = (const float*)d_in[10];
    const float* w_agg2 = (const float*)d_in[11];
    const float* b_agg2 = (const float*)d_in[12];
    const float* w_out  = (const float*)d_in[13];
    const float* b_out  = (const float*)d_in[14];

    // ---- workspace carve (byte offsets, 256B aligned) ----
    char* p = (char*)d_ws;
    size_t off = 0;
    auto carve = [&](size_t bytes) {
        void* r = p + off;
        off += (bytes + 255) & ~(size_t)255;
        return r;
    };
    __bf16* wtEnc2 = (__bf16*)carve((size_t)C_DIM * C_DIM * 2);
    __bf16* wtAgg1 = (__bf16*)carve((size_t)C_DIM * C_DIM * 2);
    __bf16* wtAgg2 = (__bf16*)carve((size_t)C_DIM * C_DIM * 2);
    __bf16* wtOut  = (__bf16*)carve((size_t)C_DIM * C_DIM * 2);
    __bf16* h1     = (__bf16*)carve((size_t)M1 * C_DIM * 2);
    float*  emb    = (float*) carve((size_t)M1 * C_DIM * 4);
    __bf16* erin   = (__bf16*)carve((size_t)MA * C_DIM * 2);
    __bf16* h2     = (__bf16*)carve((size_t)MA * C_DIM * 2);
    __bf16* embw   = (__bf16*)carve((size_t)TL * N_DIM * C_DIM * 2);
    __bf16* embT   = (__bf16*)carve((size_t)TL * C_DIM * N_DIM * 2);
    float*  pos    = (float*) carve((size_t)M1 * 2 * 4);
    float*  validf = (float*) carve((size_t)M1 * 4);
    float*  posw   = (float*) carve((size_t)TL * N_DIM * 2 * 4);
    float*  maskw  = (float*) carve((size_t)TL * N_DIM * 4);
    __bf16* wnA    = (__bf16*)carve((size_t)TL * HW * N_DIM * 2);
    __bf16* gridb  = (__bf16*)carve((size_t)MG * C_DIM * 2);
    (void)ws_size; (void)in_sizes; (void)n_in; (void)out_size;

    // 1) weight converts (f32 KxN -> bf16 transposed NxK)
    {
        int blocks = (C_DIM * C_DIM) / 256;
        wt_convert_kernel<<<blocks, 256, 0, stream>>>(w_enc2, wtEnc2, C_DIM, C_DIM);
        wt_convert_kernel<<<blocks, 256, 0, stream>>>(w_agg1, wtAgg1, C_DIM, C_DIM);
        wt_convert_kernel<<<blocks, 256, 0, stream>>>(w_agg2, wtAgg2, C_DIM, C_DIM);
        wt_convert_kernel<<<blocks, 256, 0, stream>>>(w_out,  wtOut,  C_DIM, C_DIM);
    }
    // 2) enc1 + gelu -> h1 bf16
    enc1_kernel<<<dim3(M1, C_DIM / 256), 256, 0, stream>>>(kin, w_enc1, b_enc1, h1);
    // 3) pos MLP + validity
    pos_kernel<<<M1, 128, 0, stream>>>(kin, w_pos1, b_pos1, w_pos2, b_pos2, pos, validf);
    // 4) enc2 GEMM: emb(f32) = h1 @ w_enc2 + b_enc2
    gemm_wmma_bf16<false, false><<<dim3((M1 + 63) / 64, C_DIM / 256, 1), 256, 0, stream>>>(
        h1, wtEnc2, b_enc2, emb, M1, C_DIM, C_DIM, C_DIM, C_DIM, 0, 0, 0);
    // 5) temporal mean -> erin bf16 ; e0 -> embw rows 0..31
    avg_kernel<<<(MA * C_DIM) / 256, 256, 0, stream>>>(emb, erin);
    e0_kernel<<<(N_DIM * C_DIM) / 256, 256, 0, stream>>>(emb, embw);
    // 6) pos/mask window aggregation
    posw_kernel<<<1, TL * N_DIM, 0, stream>>>(pos, validf, posw, maskw);
    // 7) agg1 GEMM (+gelu, bf16 out): h2 = gelu(erin @ w_agg1 + b_agg1)
    gemm_wmma_bf16<true, true><<<dim3((MA + 63) / 64, C_DIM / 256, 1), 256, 0, stream>>>(
        erin, wtAgg1, b_agg1, h2, MA, C_DIM, C_DIM, C_DIM, C_DIM, 0, 0, 0);
    // 8) agg2 GEMM (bf16 out directly into embw rows 32..255)
    gemm_wmma_bf16<false, true><<<dim3((MA + 63) / 64, C_DIM / 256, 1), 256, 0, stream>>>(
        h2, wtAgg2, b_agg2, embw + (size_t)N_DIM * C_DIM, MA, C_DIM, C_DIM, C_DIM, C_DIM, 0, 0, 0);
    // 9) transpose emb_w per-t for splat B operand
    embT_kernel<<<(TL * N_DIM * C_DIM) / 256, 256, 0, stream>>>(embw, embT);
    // 10) normalized gaussian weights wnA[t][hw][n]
    wn_kernel<<<MG / 256, 256, 0, stream>>>(posw, maskw, wnA);
    // 11) splat GEMM (batched over t): grid(bf16) = wnA @ embT^T
    gemm_wmma_bf16<false, true><<<dim3(HW / 64, C_DIM / 256, TL), 256, 0, stream>>>(
        wnA, embT, nullptr, gridb, HW, N_DIM, N_DIM, N_DIM, C_DIM,
        (long)HW * N_DIM, (long)C_DIM * N_DIM, (long)HW * C_DIM);
    // 12) final GEMM: out(f32) = grid @ w_out + b_out  (dominant: 236 GFLOP)
    gemm_wmma_bf16<false, false><<<dim3(MG / 64, C_DIM / 256, 1), 256, 0, stream>>>(
        gridb, wtOut, b_out, d_out, MG, C_DIM, C_DIM, C_DIM, C_DIM, 0, 0, 0);
}
